// Spec_SS1D_46686294507592
// MI455X (gfx1250) — matrix operand order repaired and checked by reading, not compiled, not run on hardware
//
#include <hip/hip_runtime.h>
#include <hip/hip_bf16.h>
#include <math.h>

// Problem constants (from reference)
#define B_SZ    8
#define L_SEQ   1024
#define D_MODEL 768
#define D_INNER 768
#define D_STATE 16
#define DT_RANK 48
#define KDIR    2
#define DT_PAD  64          // DT_RANK padded to multiple of 32
#define XD_ROWS 80          // DT_RANK + 2*D_STATE

typedef __bf16 bf16_t;
typedef __attribute__((ext_vector_type(16))) __bf16 bf16x16;
typedef __attribute__((ext_vector_type(8)))  __bf16 bf16x8;
typedef __attribute__((ext_vector_type(8)))  float  f32x8;

__device__ __forceinline__ float sigmoidf_(float x){ return 1.f/(1.f+__expf(-x)); }
__device__ __forceinline__ float siluf_(float x){ return x*sigmoidf_(x); }

// CDNA5 async global->LDS copy (tracked by ASYNCcnt). dst is the wave-relative
// LDS byte offset (= low 32 bits of the flat shared address per the aperture
// mapping), src a per-lane 64-bit global address; copies 16 bytes per lane.
__device__ __forceinline__ void async_ld_b128(unsigned int lds_dst, const void* src){
  asm volatile("global_load_async_to_lds_b128 %0, %1, off"
               :: "v"(lds_dst), "v"(src) : "memory");
}
__device__ __forceinline__ void wait_asynccnt0(){
  asm volatile("s_wait_asynccnt 0x0" ::: "memory");
}
__device__ __forceinline__ unsigned int lds_addr32(const void* p){
  return (unsigned int)(unsigned long long)p;   // flat addr[31:0] == LDS offset
}

// ---------------------------------------------------------------- conversions
__global__ void f32_to_bf16_kernel(const float* __restrict__ src,
                                   bf16_t* __restrict__ dst, int n){
  int i = blockIdx.x*256 + threadIdx.x;
  if (i < n) dst[i] = (bf16_t)src[i];
}

// dt_projs_w (K, D_INNER, DT_RANK) -> bf16 (K, D_INNER, DT_PAD), zero padded
__global__ void pad_dtw_kernel(const float* __restrict__ src,
                               bf16_t* __restrict__ dst){
  int i = blockIdx.x*256 + threadIdx.x;
  if (i >= KDIR*D_INNER*DT_PAD) return;
  int r  = i & (DT_PAD-1);
  int kd = i >> 6;                       // k*D_INNER + d
  dst[i] = (r < DT_RANK) ? (bf16_t)src[kd*DT_RANK + r] : (bf16_t)0.f;
}

// ---------------------------------------------------------------- WMMA GEMM
// C(M,N) = A(M,K) x B(K,N); A bf16 row-major. TRANSB => B given as W (N,K)
// row-major, i.e. computes A @ W^T. 64x64 tile per block, 4 waves, each wave
// a 32x32 patch = 2x2 v_wmma_f32_16x16x32_bf16 fragments. LDS keeps A as
// (m,k) and B transposed as (n,k) so every fragment is two ds_load_b128.
// TRANSB path: double-buffered GLOBAL_LOAD_ASYNC_TO_LDS_B128 pipeline.
enum { EPI_XZ=0, EPI_SILU=1, EPI_XDBL=2, EPI_DT=3, EPI_OUT=4 };

template<int EPI, bool TRANSB>
__global__ __launch_bounds__(128)
void wmma_gemm_kernel(const bf16_t* __restrict__ Abase,
                      const bf16_t* __restrict__ Bbase,
                      float*  __restrict__ out0,
                      bf16_t* __restrict__ out1,
                      int M, int N, int Kd, int lda, int ldb)
{
  __shared__ __align__(16) bf16_t As[2][64][40];   // (m, k) + pad
  __shared__ __align__(16) bf16_t Bt[2][64][40];   // (n, k) + pad (transposed!)

  const int tid  = threadIdx.x;
  const int lane = tid & 31;
  const int wave = tid >> 5;
  const int mBase = blockIdx.y * 64;
  const int nBase = blockIdx.x * 64;
  const int bk = blockIdx.z;                 // batch index = b*K + k
  const int bb = bk / KDIR, kk = bk % KDIR;

  const bf16_t* A = Abase;
  const bf16_t* B = Bbase;
  if (EPI == EPI_XDBL) { A += (size_t)kk * XD_ROWS * D_INNER;
                         B += (size_t)bb * ((size_t)D_INNER * L_SEQ); }
  if (EPI == EPI_DT)   { A += (size_t)kk * (D_INNER * DT_PAD);
                         B += (size_t)bk * (DT_PAD * L_SEQ); }

  const int wm = (wave >> 1) * 32;           // wave's M offset inside tile
  const int wn = (wave &  1) * 32;           // wave's N offset inside tile
  const int l15 = lane & 15, hlf = lane >> 4;

  f32x8 acc[2][2] = {};

  // thread -> staged row mapping (same for A and transposed-B staging)
  const int srow = tid >> 1;                 // 0..63
  const int scb  = (tid & 1) * 16;           // 0 or 16 (k offset)

  if (TRANSB) {
    // ---------------- async double-buffered pipeline ----------------
    auto issue = [&](int k0, int buf) {
      const bf16_t* sa = A + (size_t)(mBase + srow) * lda + k0 + scb;
      unsigned int da = lds_addr32(&As[buf][srow][scb]);
      async_ld_b128(da,      sa);
      async_ld_b128(da + 16, sa + 8);
      const bf16_t* sb = B + (size_t)(nBase + srow) * ldb + k0 + scb;
      unsigned int db = lds_addr32(&Bt[buf][srow][scb]);
      async_ld_b128(db,      sb);
      async_ld_b128(db + 16, sb + 8);
    };
    issue(0, 0);
    int cur = 0;
    for (int k0 = 0; k0 < Kd; k0 += 32) {
      wait_asynccnt0();          // own tile-k0 loads in LDS
      __syncthreads();           // everyone's tile-k0 loads in LDS
      if (k0 + 32 < Kd) issue(k0 + 32, cur ^ 1);   // prev readers all passed
      #pragma unroll
      for (int i = 0; i < 2; ++i) {
        bf16x16 afrag;
        {
          const bf16x8* p0 = (const bf16x8*)&As[cur][wm + i*16 + l15][hlf*16];
          const bf16x8* p1 = (const bf16x8*)&As[cur][wm + i*16 + l15][hlf*16 + 8];
          afrag = __builtin_shufflevector(*p0, *p1, 0,1,2,3,4,5,6,7,8,9,10,11,12,13,14,15);
        }
        #pragma unroll
        for (int j = 0; j < 2; ++j) {
          const bf16x8* q = (const bf16x8*)&Bt[cur][wn + j*16 + l15][hlf*16];
          bf16x16 bfrag = __builtin_shufflevector(q[0], q[1], 0,1,2,3,4,5,6,7,8,9,10,11,12,13,14,15);
          acc[i][j] = __builtin_amdgcn_wmma_f32_16x16x32_bf16(
              false, afrag, false, bfrag, (short)0, acc[i][j], false, false);
        }
      }
      cur ^= 1;
    }
  } else {
    // ---------------- guarded manual staging (M=80 / row-major B) ----------
    for (int k0 = 0; k0 < Kd; k0 += 32) {
      {
        int gm = mBase + srow;
        if (gm < M) {
          const bf16_t* src = A + (size_t)gm * lda + k0 + scb;
          #pragma unroll
          for (int j = 0; j < 16; ++j) As[0][srow][scb + j] = src[j];
        } else {
          #pragma unroll
          for (int j = 0; j < 16; ++j) As[0][srow][scb + j] = (bf16_t)0.f;
        }
      }
      {
        // B row-major (k,n): transpose on store into Bt[n][k]
        int kr = tid >> 2;             // 0..31
        int nb = (tid & 3) * 16;
        const bf16_t* src = B + (size_t)(k0 + kr) * ldb + nBase + nb;
        bf16_t tmp[16];
        #pragma unroll
        for (int j = 0; j < 16; ++j) tmp[j] = src[j];
        #pragma unroll
        for (int j = 0; j < 16; ++j) Bt[0][nb + j][kr] = tmp[j];
      }
      __syncthreads();
      #pragma unroll
      for (int i = 0; i < 2; ++i) {
        bf16x16 afrag;
        {
          const bf16x8* p0 = (const bf16x8*)&As[0][wm + i*16 + l15][hlf*16];
          const bf16x8* p1 = (const bf16x8*)&As[0][wm + i*16 + l15][hlf*16 + 8];
          afrag = __builtin_shufflevector(*p0, *p1, 0,1,2,3,4,5,6,7,8,9,10,11,12,13,14,15);
        }
        #pragma unroll
        for (int j = 0; j < 2; ++j) {
          const bf16x8* q = (const bf16x8*)&Bt[0][wn + j*16 + l15][hlf*16];
          bf16x16 bfrag = __builtin_shufflevector(q[0], q[1], 0,1,2,3,4,5,6,7,8,9,10,11,12,13,14,15);
          acc[i][j] = __builtin_amdgcn_wmma_f32_16x16x32_bf16(
              false, afrag, false, bfrag, (short)0, acc[i][j], false, false);
        }
      }
      __syncthreads();
    }
  }

  // ---- epilogue (C/D layout: VGPR r holds M=r (lanes 0-15), M=r+8 (16-31))
  #pragma unroll
  for (int i = 0; i < 2; ++i)
    #pragma unroll
    for (int j = 0; j < 2; ++j)
      #pragma unroll
      for (int r = 0; r < 8; ++r) {
        int m = mBase + wm + i*16 + hlf*8 + r;
        int n = nBase + wn + j*16 + l15;
        if (m >= M) continue;
        float v = acc[i][j][r];
        if (EPI == EPI_XZ) {                       // split: xc (bf16) | z (f32)
          if (n < D_MODEL) out1[(size_t)m*D_MODEL + n] = (bf16_t)v;
          else             out0[(size_t)m*D_MODEL + (n - D_MODEL)] = v;
        } else if (EPI == EPI_SILU) {              // u = silu(.) bf16
          out1[(size_t)m*N + n] = (bf16_t)siluf_(v);
        } else if (EPI == EPI_XDBL) {              // dt rows -> bf16; B/C -> f32
          if (m < DT_RANK)
            out1[(size_t)bk*(DT_PAD*L_SEQ) + (size_t)m*L_SEQ + n] = (bf16_t)v;
          else
            out0[(size_t)bk*(2*D_STATE*L_SEQ) + (size_t)(m - DT_RANK)*L_SEQ + n] = v;
        } else if (EPI == EPI_DT) {                // dts pre-softplus f32
          out0[(size_t)bk*((size_t)D_INNER*L_SEQ) + (size_t)m*L_SEQ + n] = v;
        } else {                                   // EPI_OUT plain f32
          out0[(size_t)m*N + n] = v;
        }
      }
}

// ---------------------------------------------------------------- selective scan
// One thread = one (b,k,d) channel, 16-state recurrence in registers.
// B_t / C_t (shared by all d of a (b,k)) staged through LDS in 64-step tiles.
// k=1 walks positions backwards (p = L-1-t) which folds both reference flips.
__global__ __launch_bounds__(256)
void scan_kernel(const float* __restrict__ dts_raw, const bf16_t* __restrict__ u_bf,
                 const float* __restrict__ BC,      const float* __restrict__ A_logs,
                 const float* __restrict__ Ds,      const float* __restrict__ dt_b,
                 float* __restrict__ y01)
{
  __shared__ float sBC[2*D_STATE][64 + 1];

  const int tid = threadIdx.x;
  const int bk = blockIdx.x / 3;               // 768/256 = 3 blocks per (b,k)
  const int d  = (blockIdx.x % 3) * 256 + tid;
  const int bb = bk / KDIR, kk = bk % KDIR;
  const int kd = kk * D_INNER + d;

  float Ar[D_STATE];
  #pragma unroll
  for (int n = 0; n < D_STATE; ++n) Ar[n] = -__expf(A_logs[kd*D_STATE + n]);
  const float Dv  = Ds[kd];
  const float dtb = dt_b[kd];

  float h[D_STATE];
  #pragma unroll
  for (int n = 0; n < D_STATE; ++n) h[n] = 0.f;

  const size_t chanbase = ((size_t)bk * D_INNER + d) * L_SEQ;
  const size_t bcbase   = (size_t)bk * (2*D_STATE) * L_SEQ;
  const size_t ubase    = (size_t)bb * ((size_t)D_INNER * L_SEQ) + (size_t)d * L_SEQ;

  for (int t0 = 0; t0 < L_SEQ; t0 += 64) {
    #pragma unroll
    for (int q = 0; q < 8; ++q) {              // 32 rows x 64 cols, 8 per thread
      int idx = tid * 8 + q;
      int row = idx >> 6, col = idx & 63;
      int t = t0 + col;
      int p = kk ? (L_SEQ - 1 - t) : t;
      sBC[row][col] = BC[bcbase + (size_t)row * L_SEQ + p];
    }
    __syncthreads();
    for (int j = 0; j < 64; ++j) {
      int t = t0 + j;
      int p = kk ? (L_SEQ - 1 - t) : t;
      float dtv   = dts_raw[chanbase + p] + dtb;
      float delta = (dtv > 20.f) ? dtv : log1pf(__expf(dtv));
      float uu    = (float)u_bf[ubase + p];
      float du    = delta * uu;
      float y = 0.f;
      #pragma unroll
      for (int n = 0; n < D_STATE; ++n) {
        float dA = __expf(delta * Ar[n]);
        h[n] = dA * h[n] + du * sBC[n][j];
        y += h[n] * sBC[D_STATE + n][j];
      }
      y01[chanbase + p] = y + Dv * uu;
    }
    __syncthreads();
  }
}

// ---------------------------------------------------------------- LN + gate
__global__ __launch_bounds__(256)
void ln_gate_kernel(const float* __restrict__ y01, const float* __restrict__ z,
                    const float* __restrict__ g,   const float* __restrict__ be,
                    bf16_t* __restrict__ yn)
{
  __shared__ float red[256];
  const int tid = threadIdx.x;
  const int bl  = blockIdx.x;
  const int bb  = bl >> 10, l = bl & (L_SEQ - 1);

  float vals[3];
  float s = 0.f;
  #pragma unroll
  for (int i = 0; i < 3; ++i) {
    int d = tid + i*256;
    size_t i0 = ((size_t)(bb*KDIR + 0) * D_INNER + d) * L_SEQ + l;
    size_t i1 = ((size_t)(bb*KDIR + 1) * D_INNER + d) * L_SEQ + l;
    float v = y01[i0] + y01[i1];
    vals[i] = v; s += v;
  }
  red[tid] = s; __syncthreads();
  for (int off = 128; off > 0; off >>= 1) {
    if (tid < off) red[tid] += red[tid + off];
    __syncthreads();
  }
  float mean = red[0] * (1.f / D_INNER);
  __syncthreads();
  float vs = 0.f;
  #pragma unroll
  for (int i = 0; i < 3; ++i) { float dv = vals[i] - mean; vs += dv*dv; }
  red[tid] = vs; __syncthreads();
  for (int off = 128; off > 0; off >>= 1) {
    if (tid < off) red[tid] += red[tid + off];
    __syncthreads();
  }
  float rstd = rsqrtf(red[0] * (1.f / D_INNER) + 1e-5f);
  size_t rowbase = (size_t)bl * D_MODEL;
  #pragma unroll
  for (int i = 0; i < 3; ++i) {
    int d = tid + i*256;
    float zv = z[rowbase + d];
    float yv = (vals[i] - mean) * rstd * g[d] + be[d];
    yn[rowbase + d] = (bf16_t)(yv * siluf_(zv));
  }
}

// ---------------------------------------------------------------- launch
extern "C" void kernel_launch(void* const* d_in, const int* in_sizes, int n_in,
                              void* d_out, int out_size, void* d_ws, size_t ws_size,
                              hipStream_t stream)
{
  (void)in_sizes; (void)n_in; (void)out_size; (void)ws_size;
  const float* x          = (const float*)d_in[0];
  const float* in_proj_w  = (const float*)d_in[1];
  const float* proj_w     = (const float*)d_in[2];
  const float* x_proj_w   = (const float*)d_in[3];
  const float* dt_projs_w = (const float*)d_in[4];
  const float* dt_projs_b = (const float*)d_in[5];
  const float* A_logs     = (const float*)d_in[6];
  const float* Ds         = (const float*)d_in[7];
  const float* ln_g       = (const float*)d_in[8];
  const float* ln_b       = (const float*)d_in[9];
  const float* out_proj_w = (const float*)d_in[10];

  char* ws = (char*)d_ws;
  size_t off = 0;
  auto alloc = [&](size_t bytes) -> char* {
    char* p = ws + off;
    off = (off + bytes + 255) & ~((size_t)255);
    return p;
  };

  const size_t ML  = (size_t)B_SZ * L_SEQ;          // 8192 rows
  bf16_t* x_bf     = (bf16_t*)alloc(ML * D_MODEL * 2);
  bf16_t* win_bf   = (bf16_t*)alloc((size_t)2*D_INNER * D_MODEL * 2);
  bf16_t* wproj_bf = (bf16_t*)alloc((size_t)D_INNER * D_INNER * 2);
  bf16_t* wout_bf  = (bf16_t*)alloc((size_t)D_MODEL * D_INNER * 2);
  bf16_t* wxp_bf   = (bf16_t*)alloc((size_t)KDIR * XD_ROWS * D_INNER * 2);
  bf16_t* wdt_bf   = (bf16_t*)alloc((size_t)KDIR * D_INNER * DT_PAD * 2);
  bf16_t* xc_bf    = (bf16_t*)alloc(ML * D_INNER * 2);
  float*  z_f      = (float*) alloc(ML * D_INNER * 4);
  bf16_t* u_bf     = (bf16_t*)alloc(ML * D_INNER * 2);
  bf16_t* dtp_bf   = (bf16_t*)alloc((size_t)B_SZ*KDIR * DT_PAD * L_SEQ * 2);
  float*  BC_f     = (float*) alloc((size_t)B_SZ*KDIR * 2*D_STATE * L_SEQ * 4);
  float*  dts_f    = (float*) alloc((size_t)B_SZ*KDIR * D_INNER * L_SEQ * 4);
  float*  y01_f    = (float*) alloc((size_t)B_SZ*KDIR * D_INNER * L_SEQ * 4);
  bf16_t* yn_bf    = (bf16_t*)alloc(ML * D_INNER * 2);

  auto cvt = [&](const float* s, bf16_t* dst, int n){
    f32_to_bf16_kernel<<<(n + 255)/256, 256, 0, stream>>>(s, dst, n);
  };
  cvt(x,          x_bf,     (int)(ML * D_MODEL));
  cvt(in_proj_w,  win_bf,   2*D_INNER * D_MODEL);
  cvt(proj_w,     wproj_bf, D_INNER * D_INNER);
  cvt(out_proj_w, wout_bf,  D_MODEL * D_INNER);
  cvt(x_proj_w,   wxp_bf,   KDIR * XD_ROWS * D_INNER);
  pad_dtw_kernel<<<(KDIR*D_INNER*DT_PAD + 255)/256, 256, 0, stream>>>(dt_projs_w, wdt_bf);

  // GEMM1: xz = x @ in_proj_w.T   -> xc (bf16) | z (f32)
  wmma_gemm_kernel<EPI_XZ, true><<<dim3((2*D_INNER)/64, ML/64, 1), 128, 0, stream>>>(
      x_bf, win_bf, z_f, xc_bf, (int)ML, 2*D_INNER, D_MODEL, D_MODEL, D_MODEL);

  // GEMM2: u = silu(xc @ proj_w.T)  (flat buffer doubles as (d,l) matrix per b)
  wmma_gemm_kernel<EPI_SILU, true><<<dim3(D_INNER/64, ML/64, 1), 128, 0, stream>>>(
      xc_bf, wproj_bf, nullptr, u_bf, (int)ML, D_INNER, D_INNER, D_INNER, D_INNER);

  (void)hipMemsetAsync(dtp_bf, 0, (size_t)B_SZ*KDIR * DT_PAD * L_SEQ * 2, stream);

  // GEMM3 (batched over b,k): x_dbl = x_proj_w[k] @ U_b  -> dt rows | B,C rows
  wmma_gemm_kernel<EPI_XDBL, false><<<dim3(L_SEQ/64, 2, B_SZ*KDIR), 128, 0, stream>>>(
      wxp_bf, u_bf, BC_f, dtp_bf, XD_ROWS, L_SEQ, D_INNER, D_INNER, L_SEQ);

  // GEMM4 (batched): dts = dt_projs_w[k] @ dt_rows (K padded 48->64)
  wmma_gemm_kernel<EPI_DT, false><<<dim3(L_SEQ/64, D_INNER/64, B_SZ*KDIR), 128, 0, stream>>>(
      wdt_bf, dtp_bf, dts_f, nullptr, D_INNER, L_SEQ, DT_PAD, DT_PAD, L_SEQ);

  // Selective scan: both directions, all (b,k,d) channels in parallel
  scan_kernel<<<(B_SZ*KDIR*D_INNER)/256, 256, 0, stream>>>(
      dts_f, u_bf, BC_f, A_logs, Ds, dt_projs_b, y01_f);

  // Combine directions + LayerNorm + silu(z) gate -> bf16 rows
  ln_gate_kernel<<<B_SZ*L_SEQ, 256, 0, stream>>>(y01_f, z_f, ln_g, ln_b, yn_bf);

  // GEMM5: out = yn @ out_proj_w.T (f32 to d_out)
  wmma_gemm_kernel<EPI_OUT, true><<<dim3(D_MODEL/64, ML/64, 1), 128, 0, stream>>>(
      yn_bf, wout_bf, (float*)d_out, nullptr, (int)ML, D_MODEL, D_INNER, D_INNER, D_INNER);
}
